// CustomLSTM_85839216378391
// MI455X (gfx1250) — matrix operand order repaired and checked by reading, not compiled, not run on hardware
//
#include <hip/hip_runtime.h>

// ---------- types ----------
typedef __attribute__((ext_vector_type(16))) __bf16        v16bf;
typedef __attribute__((ext_vector_type(8)))  float         v8f;
typedef __attribute__((ext_vector_type(4)))  unsigned int  u32x4;
typedef __attribute__((ext_vector_type(8)))  int           i32x8;
typedef __attribute__((ext_vector_type(4)))  int           i32x4;

union Frag16 { u32x4 q[2]; v16bf v; };   // 32B of bf16 = one WMMA A/B fragment per lane

// ---------- problem constants ----------
constexpr int B_  = 64;
constexpr int S_  = 512;
constexpr int I_  = 1024;
constexpr int H_  = 1024;
constexpr int G4H = 4 * H_;          // 4096
constexpr int NT  = G4H / 16;        // 256 column tiles of 16
constexpr int KB  = I_ / 32;         // 32 k-blocks of 32
constexpr int LDS_STRIDE = I_ + 8;   // 1032 elements = 2064B/row = 516 dwords (4 mod 64 banks)

__device__ __forceinline__ unsigned short f32_to_bf16(float f) {
    unsigned u = __float_as_uint(f);
    unsigned r = 0x7FFFu + ((u >> 16) & 1u);     // round-to-nearest-even
    return (unsigned short)((u + r) >> 16);
}
__device__ __forceinline__ float sigmoidf_(float x) {
    return 1.0f / (1.0f + __expf(-x));
}

// ---------- TDM: DMA a 16-row x 2048B strip into LDS with padded rows ----------
// data_size = 4B units; tile = 512 x 16 units; row stride 512 units (2048 B).
// pad_enable, pad_interval=7 (256 dwords = 1024 B), pad_amount=1 (2 dwords = 8 B)
// -> LDS row stride 2064 B == LDS_STRIDE bf16 elements; second half of each row +8 B.
__device__ __forceinline__ void tdm_load_strip16(const unsigned short* gsrc, unsigned lds_off) {
    unsigned long long ga = (unsigned long long)(uintptr_t)gsrc;
    u32x4 g0;
    g0.x = 1u;                                              // count=1 user descriptor
    g0.y = lds_off;                                         // lds_addr
    g0.z = (unsigned)(ga & 0xFFFFFFFFull);                  // global_addr[31:0]
    g0.w = (unsigned)((ga >> 32) & 0x01FFFFFFull)           // global_addr[56:32]
         | 0x80000000u;                                     // type=2 ("image")
    i32x8 g1;
    g1[0] = (2 << 16) | (1 << 20) | (7 << 22) | (1 << 25);  // data_size=4B, pad en, int=256dw, amt=2dw
    g1[1] = (int)(512u << 16);                              // tensor_dim0[15:0] = 512 units
    g1[2] = (int)(16u  << 16);                              // tensor_dim1[15:0] = 16 rows
    g1[3] = (int)(512u << 16);                              // tile_dim0 = 512 units
    g1[4] = 16;                                             // tile_dim1 = 16 rows (tile_dim2=0)
    g1[5] = 512;                                            // tensor_dim0_stride = 512 units
    g1[6] = 0;
    g1[7] = 0;
    i32x4 g2 = {0, 0, 0, 0};
    i32x4 g3 = {0, 0, 0, 0};
    i32x8 g4 = {0, 0, 0, 0, 0, 0, 0, 0};                    // 6-arg toolchain: extra group slot
    __builtin_amdgcn_tensor_load_to_lds(g0, g1, g2, g3, g4, 0);
}

// ---------- pack fp32 [K=1024, N=4096] weight into CDNA5 B-fragment layout (bf16) ----------
__global__ __launch_bounds__(256) void pack_b_frag(const float* __restrict__ src,
                                                   unsigned short* __restrict__ dst) {
    unsigned idx  = blockIdx.x * 256u + threadIdx.x;     // [0, 4194304)
    unsigned j    = idx & 15u;
    unsigned lane = (idx >> 4) & 31u;
    unsigned n    = (idx >> 9) & 255u;
    unsigned kblk = idx >> 17;
    unsigned k    = kblk * 32u + ((j & 8u) ? 16u : 0u) + ((lane & 16u) ? 8u : 0u) + (j & 7u);
    unsigned col  = n * 16u + (lane & 15u);
    dst[idx] = f32_to_bf16(src[(size_t)k * G4H + col]);
}

// ---------- fp32 -> bf16 row-major convert (for x) ----------
__global__ __launch_bounds__(256) void cvt_bf16(const float* __restrict__ src,
                                                unsigned short* __restrict__ dst) {
    size_t i = (size_t)blockIdx.x * 256 + threadIdx.x;
    dst[i] = f32_to_bf16(src[i]);
}

// ---------- big GEMM: gates_pre[r, c] = x[r,:] @ W[:,c] + b[c] ----------
__global__ __launch_bounds__(256) void gemm_xw(const unsigned short* __restrict__ xbf,
                                               const unsigned short* __restrict__ Wpk,
                                               const float* __restrict__ bias,
                                               float* __restrict__ gates) {
    __shared__ __align__(16) unsigned short As[16 * LDS_STRIDE];
    const int tid = threadIdx.x;
    const int r0  = blockIdx.x * 16;
#pragma unroll
    for (int i = 0; i < 8; ++i) {                 // stage 32 rows x 512 bf16 of x (32 KB)
        int c    = tid + i * 256;                 // 16B-chunk id
        int row  = c >> 6;
        int koff = (c & 63) << 3;
        *(u32x4*)(&As[row * LDS_STRIDE + koff]) =
            *(const u32x4*)(&xbf[(size_t)(r0 + row) * I_ + koff]);
    }
    __syncthreads();

    const int wv   = tid >> 5, lane = tid & 31;
    const int nt   = blockIdx.y * 8 + wv;         // column tile [0,256)
    const int mrow = lane & 15;
    const int klo  = (lane & 16) ? 8 : 0;         // A-fragment K sub-offset
    const float bv = bias[nt * 16 + (lane & 15)];

    v8f acc;
#pragma unroll
    for (int j = 0; j < 8; ++j) acc[j] = bv;      // bias folded into accumulator

#pragma unroll 4
    for (int kb = 0; kb < KB; ++kb) {
        Frag16 a, bf;
        const unsigned short* ap = &As[mrow * LDS_STRIDE + kb * 32 + klo];
        a.q[0] = *(const u32x4*)(ap);             // K = kb*32 + klo .. +8
        a.q[1] = *(const u32x4*)(ap + 16);        // K = kb*32 + 16 + klo .. +8
        const u32x4* bp = (const u32x4*)(&Wpk[(((size_t)kb * NT + nt) * 32 + lane) * 16]);
        bf.q[0] = bp[0];
        bf.q[1] = bp[1];
        acc = __builtin_amdgcn_wmma_f32_16x16x32_bf16(false, a.v, false, bf.v,
                                                      (short)0, acc, false, false);
    }
    const int colb = nt * 16 + (lane & 15);
    const int rb   = r0 + ((lane & 16) ? 8 : 0);
#pragma unroll
    for (int r = 0; r < 8; ++r)
        gates[(size_t)(rb + r) * G4H + colb] = acc[r];
}

// ---------- persistent recurrent kernel: 32 WGs x 8 waves, 1 h-tile per wave ----------
__global__ __launch_bounds__(256) void lstm_rec(const float* __restrict__ gates,
                                                const unsigned short* __restrict__ Upk,
                                                unsigned short* __restrict__ hbuf, // 2 x B*H bf16
                                                float* __restrict__ out,
                                                unsigned* __restrict__ bar) {
    __shared__ __align__(16) unsigned short Hs[16 * LDS_STRIDE];
    const int tid  = threadIdx.x;
    const int wg   = blockIdx.x;                  // [0,32)
    const int wv   = tid >> 5, lane = tid & 31;
    const int bt   = wg >> 3;                     // batch-row tile [0,4)
    const int ht   = ((wg & 7) << 3) + wv;        // h column tile [0,64)
    const int mrow = lane & 15;
    const int klo  = (lane & 16) ? 8 : 0;
    const int bofs = (lane & 16) ? 8 : 0;
    const int hcol = ht * 16 + (lane & 15);
    const unsigned lds_base = (unsigned)(uintptr_t)(&Hs[0]);

    v8f cacc;                                      // cell state lives in VGPRs for all 512 steps
#pragma unroll
    for (int j = 0; j < 8; ++j) cacc[j] = 0.0f;

    for (int t = 0; t < S_; ++t) {
        const unsigned short* rbuf = hbuf + (size_t)(t & 1) * (B_ * H_);
        unsigned short*       wbuf = hbuf + (size_t)((t + 1) & 1) * (B_ * H_);

        // ---- TDM: one DMA stages h rows bt*16..+16 into padded LDS ----
        if (wv == 0) {
            tdm_load_strip16(&rbuf[(size_t)bt * 16 * H_], lds_base);
            __builtin_amdgcn_s_wait_tensorcnt(0);
        }
        __syncthreads();

        v8f acc[4];                                // init accumulators with precomputed xW+b
#pragma unroll
        for (int g = 0; g < 4; ++g) {
#pragma unroll
            for (int r = 0; r < 8; ++r) {
                int b = bt * 16 + bofs + r;
                acc[g][r] = gates[((size_t)b * S_ + t) * G4H + g * H_ + hcol];
            }
        }
        if (t + 1 < S_) {                          // warm L2 for next step's pre-activations
            int b = bt * 16 + bofs;
            __builtin_prefetch(&gates[((size_t)b * S_ + (t + 1)) * G4H + hcol], 0, 1);
        }

#pragma unroll 2
        for (int kb = 0; kb < KB; ++kb) {
            Frag16 a;
            // TDM pad inserts 8B after each 1024B of data -> +4 elements for K >= 512
            const unsigned short* ap =
                &Hs[mrow * LDS_STRIDE + kb * 32 + klo + ((kb >= 16) ? 4 : 0)];
            a.q[0] = *(const u32x4*)(ap);
            a.q[1] = *(const u32x4*)(ap + 16);
#pragma unroll
            for (int g = 0; g < 4; ++g) {          // i,f,g,o tiles share the A fragment
                Frag16 bf;
                int nt = g * 64 + ht;
                const u32x4* bp =
                    (const u32x4*)(&Upk[(((size_t)kb * NT + nt) * 32 + lane) * 16]);
                bf.q[0] = bp[0];
                bf.q[1] = bp[1];
                acc[g] = __builtin_amdgcn_wmma_f32_16x16x32_bf16(false, a.v, false, bf.v,
                                                                 (short)0, acc[g], false, false);
            }
        }

#pragma unroll
        for (int j = 0; j < 8; ++j) {              // gate nonlinearities + state update
            float iv = sigmoidf_(acc[0][j]);
            float fv = sigmoidf_(acc[1][j]);
            float gv = tanhf(acc[2][j]);
            float ov = sigmoidf_(acc[3][j]);
            float cv = fv * cacc[j] + iv * gv;
            cacc[j]  = cv;
            float hv = ov * tanhf(cv);
            int b = bt * 16 + bofs + j;
            out[((size_t)b * S_ + t) * H_ + hcol] = hv;                 // hidden_seq
            wbuf[(size_t)b * H_ + hcol] = f32_to_bf16(hv);              // h for next step
            if (t == S_ - 1) {
                size_t hs = (size_t)B_ * S_ * H_;
                out[hs + (size_t)b * H_ + hcol] = hv;                   // h_t
                out[hs + (size_t)B_ * H_ + (size_t)b * H_ + hcol] = cv; // c_t
            }
        }

        // ---- device-wide barrier between timesteps ----
        __syncthreads();
        if (tid == 0) {
            __threadfence();
            __hip_atomic_fetch_add(bar, 1u, __ATOMIC_RELEASE, __HIP_MEMORY_SCOPE_AGENT);
            unsigned target = 32u * (unsigned)(t + 1);
            while (__hip_atomic_load(bar, __ATOMIC_ACQUIRE, __HIP_MEMORY_SCOPE_AGENT) < target)
                __builtin_amdgcn_s_sleep(2);
        }
        __syncthreads();
    }
}

// ---------- host launcher ----------
extern "C" void kernel_launch(void* const* d_in, const int* in_sizes, int n_in,
                              void* d_out, int out_size, void* d_ws, size_t ws_size,
                              hipStream_t stream) {
    const float* x = (const float*)d_in[0];   // [64,512,1024]
    const float* W = (const float*)d_in[1];   // [1024,4096]
    const float* U = (const float*)d_in[2];   // [1024,4096]
    const float* b = (const float*)d_in[3];   // [4096]
    float* out = (float*)d_out;

    // workspace layout
    char* ws = (char*)d_ws;
    float* gates_pre = (float*)ws;                               // 64*512*4096 f32 = 512 MB
    size_t off = (size_t)B_ * S_ * G4H * sizeof(float);
    unsigned short* xbf = (unsigned short*)(ws + off);           // 64 MB
    off += (size_t)B_ * S_ * I_ * sizeof(unsigned short);
    unsigned short* Wpk = (unsigned short*)(ws + off);           // 8 MB
    off += (size_t)I_ * G4H * sizeof(unsigned short);
    unsigned short* Upk = (unsigned short*)(ws + off);           // 8 MB
    off += (size_t)I_ * G4H * sizeof(unsigned short);
    unsigned short* hbuf = (unsigned short*)(ws + off);          // 2 x 128 KB (double buffer)
    off += (size_t)2 * B_ * H_ * sizeof(unsigned short);
    unsigned* bar = (unsigned*)(ws + off);
    off += 256;

    // reset recurrent state buffers (graph-capturable stream memsets)
    (void)hipMemsetAsync(hbuf, 0, (size_t)2 * B_ * H_ * sizeof(unsigned short), stream);
    (void)hipMemsetAsync(bar, 0, 256, stream);

    // 1) convert x to bf16
    cvt_bf16<<<(B_ * S_ * I_) / 256, 256, 0, stream>>>(x, xbf);
    // 2) pack W, U into WMMA B-fragment layout
    pack_b_frag<<<(I_ * G4H) / 256, 256, 0, stream>>>(W, Wpk);
    pack_b_frag<<<(I_ * G4H) / 256, 256, 0, stream>>>(U, Upk);
    // 3) big input-projection GEMM (+bias)
    dim3 gg((B_ * S_) / 16, G4H / 128);
    gemm_xw<<<gg, 256, 0, stream>>>(xbf, Wpk, b, gates_pre);
    // 4) persistent recurrent kernel
    lstm_rec<<<32, 256, 0, stream>>>(gates_pre, Upk, hbuf, out, bar);
}
